// SingleSiteMinimizer_91285234909700
// MI455X (gfx1250) — matrix operand order repaired and checked by reading, not compiled, not run on hardware
//
#include <hip/hip_runtime.h>

typedef __bf16 bf16;
typedef __attribute__((ext_vector_type(16))) __bf16 bf16x16;
typedef __attribute__((ext_vector_type(8)))  __bf16 bf16x8;
typedef __attribute__((ext_vector_type(8)))  float  f32x8;

#define NELEM_A 8388608   // 512*32*512  (also env_L, env_R flat size)

// ---------------------------------------------------------------------------
// WMMA helpers (CDNA5 wave32, v_wmma_f32_16x16x32_bf16)
// ---------------------------------------------------------------------------
__device__ __forceinline__ f32x8 wmma_bf16(bf16x16 a, bf16x16 b, f32x8 c) {
  return __builtin_amdgcn_wmma_f32_16x16x32_bf16(false, a, false, b,
                                                 (short)0, c, false, false);
}

// A-fragment: 16x32 bf16 tile from row-major src. Lane L: row m = L%16,
// half h = L/16 holds K = {8h..8h+7} in regs 0..7 and {16+8h..} in 8..15.
__device__ __forceinline__ bf16x16 frag_a(const bf16* __restrict__ src, int ld,
                                          int row0, int k0) {
  int lane = threadIdx.x & 31;
  int m = lane & 15, h = lane >> 4;
  const bf16* p = src + (size_t)(row0 + m) * ld + k0 + h * 8;
  bf16x8 lo = *(const bf16x8*)p;
  bf16x8 hi = *(const bf16x8*)(p + 16);
  bf16x16 f;
#pragma unroll
  for (int e = 0; e < 8; ++e) { f[e] = lo[e]; f[e + 8] = hi[e]; }
  return f;
}

// B-fragment: 32x16 bf16 tile, operand stored TRANSPOSED row-major (Bt[n][k]).
// Lane L: col n = L%16, half h = L/16 holds K = {8h..8h+7} in regs 0..7 and {16+8h..} in 8..15.
__device__ __forceinline__ bf16x16 frag_bt(const bf16* __restrict__ bt, int ld,
                                           int col0, int k0) {
  int lane = threadIdx.x & 31;
  int n = lane & 15, h = lane >> 4;
  const bf16* p = bt + (size_t)(col0 + n) * ld + k0 + h * 8;
  bf16x8 lo = *(const bf16x8*)p;
  bf16x8 hi = *(const bf16x8*)(p + 16);
  bf16x16 f;
#pragma unroll
  for (int e = 0; e < 8; ++e) { f[e] = lo[e]; f[e + 8] = hi[e]; }
  return f;
}

// Async global->LDS copy, 16B per lane, ASYNCcnt-tracked (CDNA5).
// lds_addr: low 32 bits of a flat pointer into LDS == LDS byte offset.
__device__ __forceinline__ void async_b128(unsigned lds_addr, const void* g) {
  asm volatile("global_load_async_to_lds_b128 %0, %1, off"
               :: "v"(lds_addr), "v"(g) : "memory");
}
__device__ __forceinline__ void wait_async_le4() {
  asm volatile("s_wait_asynccnt 4" ::: "memory");
}
__device__ __forceinline__ void wait_async_0() {
  asm volatile("s_wait_asynccnt 0" ::: "memory");
}

// ---------------------------------------------------------------------------
// Prep kernels
// ---------------------------------------------------------------------------
__global__ void zero_acc_kernel(float* acc) {
  if (threadIdx.x < 8) acc[threadIdx.x] = 0.0f;
}

__global__ void norm_kernel(const float* __restrict__ a, float* acc) {
  int i = blockIdx.x * blockDim.x + threadIdx.x;
  int stride = gridDim.x * blockDim.x;
  float s = 0.0f;
  for (; i < NELEM_A; i += stride) { float v = a[i]; s += v * v; }
#pragma unroll
  for (int off = 16; off; off >>= 1) s += __shfl_xor(s, off, 32);
  if ((threadIdx.x & 31) == 0) atomicAdd(acc, s);
}

// ops[c][a][b], c in {I, X, P, XX, PP}; d = 32. One block of 1024 threads.
__global__ void build_ops_kernel(const float* __restrict__ theta,
                                 float* __restrict__ ops) {
  __shared__ float Xs[1024], Ps[1024];
  int t = threadIdx.x;
  int a = t >> 5, b = t & 31;
  const float inv2 = 0.70710678118654752f;
  float x0 = 0.0f, p0 = 0.0f;
  if (b == a + 1) { float v = sqrtf((float)(a + 1)) * inv2; x0 += v; p0 += v; }
  if (a == b + 1) { float v = sqrtf((float)(b + 1)) * inv2; x0 += v; p0 -= v; }
  float th = theta[0], cc = cosf(th), ss = sinf(th);
  float X = cc * x0 + ss * p0;
  float P = -ss * x0 + cc * p0;
  Xs[t] = X; Ps[t] = P;
  ops[0 * 1024 + t] = (a == b) ? 1.0f : 0.0f;
  ops[1 * 1024 + t] = X;
  ops[2 * 1024 + t] = P;
  __syncthreads();
  float xx = 0.0f, pp = 0.0f;
#pragma unroll 8
  for (int m = 0; m < 32; ++m) {
    xx += Xs[a * 32 + m] * Xs[m * 32 + b];
    pp += Ps[a * 32 + m] * Ps[m * 32 + b];
  }
  ops[3 * 1024 + t] = xx;
  ops[4 * 1024 + t] = pp;
}

// Wmat[k][(P*8+v)][(w*32+p)] = sum_c coeffs[k,w,v,c] * ops[c][P][p]   (bf16)
__global__ void build_w_kernel(const float* __restrict__ coeffs,
                               const float* __restrict__ ops,
                               bf16* __restrict__ wmat) {
  int idx = blockIdx.x * blockDim.x + threadIdx.x;   // 4*256*256 = 262144
  int k = idx >> 16;
  int rem = idx & 65535;
  int m = rem >> 8;        // P*8+v
  int col = rem & 255;     // w*32+p
  int P = m >> 3, v = m & 7, w = col >> 5, p = col & 31;
  float s = 0.0f;
#pragma unroll
  for (int c = 0; c < 5; ++c)
    s += coeffs[((k * 8 + w) * 8 + v) * 5 + c] * ops[c * 1024 + P * 32 + p];
  wmat[idx] = (bf16)s;
}

// A_T[(pr)][l] = (bf16) A[l][pr]  via 64x64 LDS tile transpose.
// grid: (512/64) * (16384/64) = 2048 blocks x 256 threads
__global__ __launch_bounds__(256)
void conv_at_kernel(const float* __restrict__ a, bf16* __restrict__ at) {
  __shared__ alignas(16) bf16 s[64 * 64];   // s[pj][li]
  int lb  = (blockIdx.x >> 8) * 64;         // l  base (8 groups)
  int prb = (blockIdx.x & 255) * 64;        // pr base (256 groups)
  int t = threadIdx.x;
#pragma unroll
  for (int e = 0; e < 16; ++e) {            // coalesced f32 reads
    int i = e * 256 + t;                    // 4096 elements
    int li = i >> 6, pj = i & 63;
    s[pj * 64 + li] = (bf16)a[(size_t)(lb + li) * 16384 + prb + pj];
  }
  __syncthreads();
#pragma unroll
  for (int e = 0; e < 2; ++e) {             // 16B bf16 writes
    int c = e * 256 + t;                    // 512 chunks of 8
    int pj = c >> 3, lg = (c & 7) * 8;
    bf16x8 v = *(const bf16x8*)&s[pj * 64 + lg];
    *(bf16x8*)&at[(size_t)(prb + pj) * 512 + lb + lg] = v;
  }
}

__global__ void conv_env_kernel(const float* __restrict__ eL,
                                const float* __restrict__ eR,
                                bf16* __restrict__ eLb,
                                bf16* __restrict__ eRb) {
  int i = blockIdx.x * blockDim.x + threadIdx.x;
  int stride = gridDim.x * blockDim.x;
  for (; i < NELEM_A; i += stride) {
    eLb[i] = (bf16)eL[i];
    eRb[i] = (bf16)eR[i];
  }
}

// ---------------------------------------------------------------------------
// GEMM1: T1[(L,w),(p,r)] = envL_k[(L,w),l] x A[l,(p,r)]   (chunk of 64 L)
// Block = 8 waves sharing one (m,r) range, covering 8 consecutive p values;
// output tile staged in LDS, then written as 16B runs (contiguous in p) into
// T1t[Lloc][r][w*32+p] (Bt layout for GEMM2).
// grid: 16 (mt) * 4 (pg) * 16 (rg) = 1024 blocks x 256 threads
// ---------------------------------------------------------------------------
__global__ __launch_bounds__(256)
void gemm1_kernel(const bf16* __restrict__ envLk,  // [4096,512] rows = L*8+w
                  const bf16* __restrict__ aT,     // [16384,512] (Bt)
                  bf16* __restrict__ t1t,          // [64][512][256]
                  int Lb) {
  __shared__ alignas(16) bf16 sbuf[32 * 32 * 8];   // [ml][rl][pj]
  int wave = threadIdx.x >> 5, lane = threadIdx.x & 31;
  int mt = blockIdx.x >> 6;          // 0..15
  int pg = (blockIdx.x >> 4) & 3;    // 0..3
  int rg = blockIdx.x & 15;          // 0..15
  int mloc0 = mt * 32;
  int row0 = Lb * 8 + mloc0;                       // global row into envL_k
  int col0 = (pg * 8 + wave) * 512 + rg * 32;      // this wave's p, r range
  f32x8 c00 = {}, c01 = {}, c10 = {}, c11 = {};
  for (int k0 = 0; k0 < 512; k0 += 32) {
    bf16x16 a0 = frag_a(envLk, 512, row0, k0);
    bf16x16 a1 = frag_a(envLk, 512, row0 + 16, k0);
    bf16x16 b0 = frag_bt(aT, 512, col0, k0);
    bf16x16 b1 = frag_bt(aT, 512, col0 + 16, k0);
    c00 = wmma_bf16(a0, b0, c00);
    c01 = wmma_bf16(a0, b1, c01);
    c10 = wmma_bf16(a1, b0, c10);
    c11 = wmma_bf16(a1, b1, c11);
  }
  int h = lane >> 4, nl = lane & 15;
  const f32x8 cs[4] = {c00, c01, c10, c11};
#pragma unroll
  for (int q = 0; q < 4; ++q) {
    int mb = (q >> 1) * 16;
    int rl = (q & 1) * 16 + nl;
#pragma unroll
    for (int i = 0; i < 8; ++i) {
      int ml = mb + h * 8 + i;
      sbuf[(ml * 32 + rl) * 8 + wave] = (bf16)cs[q][i];
    }
  }
  __syncthreads();
#pragma unroll
  for (int it = 0; it < 4; ++it) {                 // 1024 chunks of 16B
    int c = it * 256 + threadIdx.x;
    int ml = c >> 5, rl = c & 31;
    bf16x8 v = *(const bf16x8*)&sbuf[(ml * 32 + rl) * 8];
    int m = mloc0 + ml;
    int Lloc = m >> 3, w = m & 7;
    int r = rg * 32 + rl;
    *(bf16x8*)&t1t[((size_t)Lloc * 512 + r) * 256 + w * 32 + pg * 8] = v;
  }
}

// ---------------------------------------------------------------------------
// GEMM2: per Lloc, T2tile = Wmat_k[256,256] x T1[Lloc][wp=256, r=512]
// stored as T2[(Lloc*32+P)][(v*512+r)] (bf16) = row-major A for GEMM3.
// Lane-consecutive r -> stores coalesce. grid: 1024 blocks.
// ---------------------------------------------------------------------------
__global__ __launch_bounds__(256)
void gemm2_kernel(const bf16* __restrict__ wmatk,  // [256,256]
                  const bf16* __restrict__ t1t,    // [64][512][256]
                  bf16* __restrict__ t2) {         // [2048,4096]
  int id = blockIdx.x * 8 + (threadIdx.x >> 5);
  int Lloc = id >> 7;
  int rem = id & 127;
  int tm = rem >> 4;   // 0..7   (M = 256)
  int tn = rem & 15;   // 0..15  (N = 512)
  const bf16* bt = t1t + (size_t)Lloc * 512 * 256;
  int row0 = tm * 32, col0 = tn * 32;
  f32x8 c00 = {}, c01 = {}, c10 = {}, c11 = {};
  for (int k0 = 0; k0 < 256; k0 += 32) {
    bf16x16 a0 = frag_a(wmatk, 256, row0, k0);
    bf16x16 a1 = frag_a(wmatk, 256, row0 + 16, k0);
    bf16x16 b0 = frag_bt(bt, 256, col0, k0);
    bf16x16 b1 = frag_bt(bt, 256, col0 + 16, k0);
    c00 = wmma_bf16(a0, b0, c00);
    c01 = wmma_bf16(a0, b1, c01);
    c10 = wmma_bf16(a1, b0, c10);
    c11 = wmma_bf16(a1, b1, c11);
  }
  int lane = threadIdx.x & 31;
  int h = lane >> 4, nl = lane & 15;
  const f32x8 cs[4] = {c00, c01, c10, c11};
#pragma unroll
  for (int q = 0; q < 4; ++q) {
    int mb = row0 + (q >> 1) * 16;
    int r = col0 + (q & 1) * 16 + nl;
#pragma unroll
    for (int i = 0; i < 8; ++i) {
      int m = mb + h * 8 + i;           // P*8+v
      int P = m >> 3, v = m & 7;
      t2[(size_t)(Lloc * 32 + P) * 4096 + v * 512 + r] = (bf16)cs[q][i];
    }
  }
}

// ---------------------------------------------------------------------------
// GEMM3 + fused epilogue. 8 waves share one 32-col B range; B tile (32x32)
// double-buffered in LDS via global_load_async_to_lds_b128 (ASYNCcnt).
// acc_k += sum( (T2 x envR^T) .* mps ).  grid: 8 (mt) * 16 (tn) = 128 blocks.
// ---------------------------------------------------------------------------
__global__ __launch_bounds__(256)
void gemm3_kernel(const bf16* __restrict__ t2,    // [2048,4096]
                  const bf16* __restrict__ envRk, // [512,4096] (Bt)
                  const float* __restrict__ mps,  // viewed [16384,512]
                  float* __restrict__ acc_k,
                  int Lb) {
  __shared__ alignas(16) bf16 bbuf[2][32 * 32];   // [n_local][k_local]
  int wave = threadIdx.x >> 5, lane = threadIdx.x & 31;
  int mt = blockIdx.x >> 4;    // 0..7
  int tn = blockIdx.x & 15;    // 0..15
  int col0 = tn * 32;
  int row0 = mt * 256 + wave * 32;

  if (wave == 0) {             // prologue: buffer 0 <- K-step 0
#pragma unroll
    for (int c = 0; c < 4; ++c)
      async_b128((unsigned)(size_t)&bbuf[0][lane * 32 + c * 8],
                 envRk + (size_t)(col0 + lane) * 4096 + c * 8);
  }

  f32x8 c00 = {}, c01 = {}, c10 = {}, c11 = {};
  for (int kk = 0; kk < 128; ++kk) {
    int k0 = kk * 32;
    if (wave == 0) {
      if (kk + 1 < 128) {
#pragma unroll
        for (int c = 0; c < 4; ++c)
          async_b128((unsigned)(size_t)&bbuf[(kk + 1) & 1][lane * 32 + c * 8],
                     envRk + (size_t)(col0 + lane) * 4096 + k0 + 32 + c * 8);
        wait_async_le4();      // in-order: current buffer's 4 are complete
      } else {
        wait_async_0();
      }
    }
    __syncthreads();
    const bf16* bt = (const bf16*)&bbuf[kk & 1][0];  // [32][32], ld = 32
    bf16x16 b0 = frag_bt(bt, 32, 0, 0);
    bf16x16 b1 = frag_bt(bt, 32, 16, 0);
    bf16x16 a0 = frag_a(t2, 4096, row0, k0);
    bf16x16 a1 = frag_a(t2, 4096, row0 + 16, k0);
    c00 = wmma_bf16(a0, b0, c00);
    c01 = wmma_bf16(a0, b1, c01);
    c10 = wmma_bf16(a1, b0, c10);
    c11 = wmma_bf16(a1, b1, c11);
    __syncthreads();           // all reads done before next overwrite
  }

  int h = lane >> 4, nl = lane & 15;
  const f32x8 cs[4] = {c00, c01, c10, c11};
  float local = 0.0f;
#pragma unroll
  for (int q = 0; q < 4; ++q) {
    int mb = row0 + (q >> 1) * 16;
    int n = col0 + (q & 1) * 16 + nl;
#pragma unroll
    for (int i = 0; i < 8; ++i) {
      int m = mb + h * 8 + i;                     // (Lloc*32 + P)
      local += cs[q][i] * mps[(size_t)(Lb * 32 + m) * 512 + n];
    }
  }
#pragma unroll
  for (int off = 16; off; off >>= 1) local += __shfl_xor(local, off, 32);
  if (lane == 0) atomicAdd(acc_k, local);
}

__global__ void finalize_kernel(const float* __restrict__ pref,
                                const float* __restrict__ acc,
                                float* __restrict__ out) {
  if (threadIdx.x == 0) {
    float S = acc[0];
    float o = 0.0f;
#pragma unroll
    for (int k = 0; k < 4; ++k) o += pref[k] * acc[1 + k];
    out[0] = o / S;   // the two An normalizations fold into 1/sum(A*A)
  }
}

// ---------------------------------------------------------------------------
extern "C" void kernel_launch(void* const* d_in, const int* in_sizes, int n_in,
                              void* d_out, int out_size, void* d_ws,
                              size_t ws_size, hipStream_t stream) {
  (void)in_sizes; (void)n_in; (void)out_size; (void)ws_size;
  const float* mps    = (const float*)d_in[0];
  const float* theta  = (const float*)d_in[1];
  const float* envL   = (const float*)d_in[2];
  const float* envR   = (const float*)d_in[3];
  const float* coeffs = (const float*)d_in[4];
  const float* pref   = (const float*)d_in[5];
  float* out = (float*)d_out;

  char* ws = (char*)d_ws;
  const size_t MiB = 1024 * 1024;
  float* acc  = (float*)(ws);                 // [8] scalars
  float* ops  = (float*)(ws + 1024);          // 5*32*32 f32
  bf16*  wmat = (bf16*) (ws + 64 * 1024);     // 4*256*256 bf16 (512 KiB)
  bf16*  aT   = (bf16*) (ws + 1 * MiB);       // 16 MiB
  bf16*  eLb  = (bf16*) (ws + 17 * MiB);      // 16 MiB
  bf16*  eRb  = (bf16*) (ws + 33 * MiB);      // 16 MiB
  bf16*  t1t  = (bf16*) (ws + 49 * MiB);      // 16 MiB (per-chunk, reused)
  bf16*  t2   = (bf16*) (ws + 65 * MiB);      // 16 MiB (per-chunk, reused)

  zero_acc_kernel<<<1, 32, 0, stream>>>(acc);
  norm_kernel<<<1024, 256, 0, stream>>>(mps, acc);
  build_ops_kernel<<<1, 1024, 0, stream>>>(theta, ops);
  build_w_kernel<<<1024, 256, 0, stream>>>(coeffs, ops, wmat);
  conv_at_kernel<<<2048, 256, 0, stream>>>(mps, aT);
  conv_env_kernel<<<2048, 256, 0, stream>>>(envL, envR, eLb, eRb);

  for (int k = 0; k < 4; ++k) {
    const bf16* eLk = eLb + (size_t)k * 4096 * 512;
    const bf16* eRk = eRb + (size_t)k * 512 * 4096;
    const bf16* wk  = wmat + (size_t)k * 256 * 256;
    for (int cb = 0; cb < 8; ++cb) {
      int Lb = cb * 64;   // chunk of 64 L values
      gemm1_kernel<<<1024, 256, 0, stream>>>(eLk, aT, t1t, Lb);
      gemm2_kernel<<<1024, 256, 0, stream>>>(wk, t1t, t2);
      gemm3_kernel<<<128, 256, 0, stream>>>(t2, eRk, mps, acc + 1 + k, Lb);
    }
  }
  finalize_kernel<<<1, 1, 0, stream>>>(pref, acc, out);
}
